// ParticleINN_80607946211964
// MI455X (gfx1250) — matrix operand order repaired and checked by reading, not compiled, not run on hardware
//
#include <hip/hip_runtime.h>
#include <cstddef>

// ---------------------------------------------------------------------------
// Types for CDNA5 WMMA (wave32, v_wmma_f32_16x16x32_bf16)
// ---------------------------------------------------------------------------
typedef __attribute__((ext_vector_type(16))) __bf16 v16bf;
typedef __attribute__((ext_vector_type(8)))  float  v8f;

struct __attribute__((aligned(16))) I4 { int x, y, z, w; };
union FragU { v16bf v; I4 i[2]; };

#define L2PI_F 0.9189385332046727f

// Block shape: 64 rows (4 WMMA M-tiles) per 256-thread block (8 waves).
// Each wave owns 2 adjacent N-tiles (layers 1-2) over all 4 M-tiles, so each
// B-fragment load feeds 4 WMMAs and weight traffic from L2 is amortized 4x
// vs. a 16-row block.
#define MROWS 64
#define ASTRIDE 264   // bf16 elements per LDS activation row (256 + 8 pad)

__device__ __forceinline__ unsigned short f2bf(float f) {
  unsigned u = __float_as_uint(f);
  u += 0x7FFFu + ((u >> 16) & 1u);            // round-to-nearest-even
  return (unsigned short)(u >> 16);
}

// A-fragment (16x32 bf16, M x K): lanes 0-15 -> M=lane, K {k0+0..7, k0+16..23};
// lanes 16-31 -> K {k0+8..15, k0+24..31}. aRow already includes m*stride+hi*8.
__device__ __forceinline__ v16bf load_a_frag(const unsigned short* aRow, int k0) {
  FragU f;
  f.i[0] = *(const I4*)(aRow + k0);
  f.i[1] = *(const I4*)(aRow + k0 + 16);
  return f.v;
}
// B-fragment (32x16 bf16, K x N) from N-major (transposed) weights:
// lane holds column N=lane&15, K {k0+(lane>>4)*8+0..7, k0+16+(lane>>4)*8+0..7}.
__device__ __forceinline__ v16bf load_b_frag(const unsigned short* bRow, int k0) {
  FragU f;
  f.i[0] = *(const I4*)(bRow + k0);
  f.i[1] = *(const I4*)(bRow + k0 + 16);
  return f.v;
}

// MT m-tiles x NT n-tiles per wave; B fragments reused across all m-tiles.
template <int K, int KROW, int MT, int NT>
__device__ __forceinline__ void gemm_block(const unsigned short* sIn,
                                           const unsigned short* Wt,
                                           int n0, int lane,
                                           v8f acc[MT][NT]) {
  const int m  = lane & 15;
  const int hi = lane >> 4;
  const unsigned short* aBase = sIn + m * ASTRIDE + hi * 8;
  const unsigned short* bRow[NT];
#pragma unroll
  for (int nt = 0; nt < NT; ++nt)
    bRow[nt] = Wt + (size_t)(n0 + nt * 16 + m) * KROW + hi * 8;
#pragma unroll
  for (int mt = 0; mt < MT; ++mt)
#pragma unroll
    for (int nt = 0; nt < NT; ++nt)
      acc[mt][nt] = (v8f){0.f, 0.f, 0.f, 0.f, 0.f, 0.f, 0.f, 0.f};
#pragma unroll
  for (int k0 = 0; k0 < K; k0 += 32) {
    v16bf b[NT];
#pragma unroll
    for (int nt = 0; nt < NT; ++nt) b[nt] = load_b_frag(bRow[nt], k0);
#pragma unroll
    for (int mt = 0; mt < MT; ++mt) {
      v16bf a = load_a_frag(aBase + mt * 16 * ASTRIDE, k0);
#pragma unroll
      for (int nt = 0; nt < NT; ++nt)
        acc[mt][nt] = __builtin_amdgcn_wmma_f32_16x16x32_bf16(
            false, a, false, b[nt], (short)0, acc[mt][nt], false, false);
    }
  }
}

// C/D layout: lane holds N=lane&15; VGPR r holds M = r + 8*(lane>>4).
__device__ __forceinline__ void store_relu_bf16(unsigned short* lds, const v8f& acc,
                                                int mrow0, int n0,
                                                const float* bias, int lane) {
  const int n  = lane & 15;
  const int hi = lane >> 4;
  const int Ng = n0 + n;
  const float b = bias[Ng];
#pragma unroll
  for (int r = 0; r < 8; ++r) {
    float v = fmaxf(acc[r] + b, 0.f);
    lds[(mrow0 + r + 8 * hi) * ASTRIDE + Ng] = f2bf(v);
  }
}

// ---------------------------------------------------------------------------
// Per-row rational-quadratic spline (matches reference rqs_forward)
// ---------------------------------------------------------------------------
__device__ void rqs_row(const float* th, float xv, float bound, int periodic,
                        float& zOut, float& ldOut) {
  const int nb = 32;
  const float left = -bound, right = bound;
  const float minw = 1e-5f;

  float cw[33], ch[33], dv[33];

  // widths: softmax(th[0:32]) scaled, cumsum -> knot x positions
  {
    float mx = th[0];
    for (int i = 1; i < nb; ++i) mx = fmaxf(mx, th[i]);
    float e[32], s = 0.f;
    for (int i = 0; i < nb; ++i) { e[i] = __expf(th[i] - mx); s += e[i]; }
    float inv = 1.f / s, a = 0.f;
    cw[0] = left;
    for (int i = 0; i < nb; ++i) {
      a += minw + (1.f - minw * nb) * (e[i] * inv);
      cw[i + 1] = left + (right - left) * a;
    }
    cw[nb] = right;
  }
  // heights
  {
    const float* t2 = th + 32;
    float mx = t2[0];
    for (int i = 1; i < nb; ++i) mx = fmaxf(mx, t2[i]);
    float e[32], s = 0.f;
    for (int i = 0; i < nb; ++i) { e[i] = __expf(t2[i] - mx); s += e[i]; }
    float inv = 1.f / s, a = 0.f;
    ch[0] = left;
    for (int i = 0; i < nb; ++i) {
      a += minw + (1.f - minw * nb) * (e[i] * inv);
      ch[i + 1] = left + (right - left) * a;
    }
    ch[nb] = right;
  }
  // derivatives: MIN_D + softplus(th[64:97])
  for (int i = 0; i <= nb; ++i) {
    float u = th[64 + i];
    dv[i] = 1e-5f + ((u > 20.f) ? u : log1pf(__expf(u)));
  }
  if (periodic) dv[nb] = dv[0];

  float xin; bool inside;
  if (periodic) {
    float span = right - left;
    float t = fmodf(xv - left, span);
    if (t < 0.f) t += span;
    xin = left + t;
    inside = true;
  } else {
    inside = (xv >= left) && (xv <= right);
    xin = fminf(fmaxf(xv, left), right);
  }

  int idx = 0;
  for (int k = 1; k <= nb; ++k) idx += (xin >= cw[k]) ? 1 : 0;
  if (idx > nb - 1) idx = nb - 1;

  float wk = cw[idx + 1] - cw[idx];
  float hk = ch[idx + 1] - ch[idx];
  float dk = dv[idx], dk1 = dv[idx + 1];
  float delta = hk / wk;
  float t  = (xin - cw[idx]) / wk;
  float t1 = t * (1.f - t);
  float denom = delta + (dk1 + dk - 2.f * delta) * t1;
  float out = ch[idx] + hk * (delta * t * t + dk * t1) / denom;
  float omt = 1.f - t;
  float dnum = delta * delta * (dk1 * t * t + 2.f * delta * t1 + dk * omt * omt);
  float ld = __logf(dnum) - 2.f * __logf(denom);

  zOut  = inside ? out : xv;
  ldOut = inside ? ld : 0.f;
}

// ---------------------------------------------------------------------------
// Fused MLP (3 layers via WMMA) + RQS, 64 rows per block, 8 waves.
// ---------------------------------------------------------------------------
// Second LDS buffer doubles as theta storage (H1 is dead by layer 3).
union __attribute__((aligned(16))) UB {
  unsigned short h[MROWS * ASTRIDE];
  float f[MROWS * 112];
};

__global__ __launch_bounds__(256)
void mlp_rqs_kernel(const float* __restrict__ c,
                    const float* __restrict__ s0, int st0,
                    const float* __restrict__ s1, int st1,
                    const float* __restrict__ s2, int st2,
                    int nspecial,
                    const float* __restrict__ coord, int stc,
                    const unsigned short* __restrict__ W1t,
                    const unsigned short* __restrict__ W2t,
                    const unsigned short* __restrict__ W3t,
                    const float* __restrict__ b1,
                    const float* __restrict__ b2,
                    const float* __restrict__ b3,
                    float bound, int periodic, int gauss,
                    float* __restrict__ zout,
                    float* __restrict__ probout) {
  __shared__ __attribute__((aligned(16))) unsigned short sA[MROWS * ASTRIDE];
  __shared__ UB sB;

  const int tid  = threadIdx.x;
  const int lane = tid & 31;
  const int wave = tid >> 5;
  const int row0 = blockIdx.x * MROWS;

  // Build bf16 input tile [64][96] (special cols | 64 c cols | zero pad)
  for (int i = tid; i < MROWS * 96; i += 256) {
    int m = i / 96, k = i % 96;
    int r = row0 + m;
    float v = 0.f;
    if (k < nspecial) {
      v = (k == 0) ? s0[(size_t)r * st0]
        : (k == 1) ? s1[(size_t)r * st1]
                   : s2[(size_t)r * st2];
    } else if (k < nspecial + 64) {
      v = c[(size_t)r * 64 + (k - nspecial)];
    }
    sA[m * ASTRIDE + k] = f2bf(v);
  }
  __syncthreads();

  v8f acc[4][2];
  // Layer 1: [64x96] x [96x256] -> H1 in sB
  gemm_block<96, 96, 4, 2>(sA, W1t, wave * 32, lane, acc);
#pragma unroll
  for (int mt = 0; mt < 4; ++mt) {
    store_relu_bf16(sB.h, acc[mt][0], mt * 16, wave * 32,      b1, lane);
    store_relu_bf16(sB.h, acc[mt][1], mt * 16, wave * 32 + 16, b1, lane);
  }
  __syncthreads();

  // Layer 2: [64x256] x [256x256] -> H2 in sA
  gemm_block<256, 256, 4, 2>(sB.h, W2t, wave * 32, lane, acc);
#pragma unroll
  for (int mt = 0; mt < 4; ++mt) {
    store_relu_bf16(sA, acc[mt][0], mt * 16, wave * 32,      b2, lane);
    store_relu_bf16(sA, acc[mt][1], mt * 16, wave * 32 + 16, b2, lane);
  }
  __syncthreads();

  // Layer 3: [64x256] x [256x112] (97 real cols) -> theta f32 aliased in sB
  if (wave < 7) {
    v8f acc3[4][1];
    gemm_block<256, 256, 4, 1>(sA, W3t, wave * 16, lane, acc3);
    const int n  = lane & 15;
    const int hi = lane >> 4;
    const int Ng = wave * 16 + n;
    const float bb = b3[Ng];
#pragma unroll
    for (int mt = 0; mt < 4; ++mt)
#pragma unroll
      for (int r = 0; r < 8; ++r)
        sB.f[(mt * 16 + r + 8 * hi) * 112 + Ng] = acc3[mt][0][r] + bb;
  }
  __syncthreads();

  // RQS per row (64 rows, threads 0..63)
  if (tid < MROWS) {
    int r = row0 + tid;
    float xv = coord[(size_t)r * stc];
    float z, ld;
    rqs_row(&sB.f[tid * 112], xv, bound, periodic, z, ld);
    if (zout) zout[r] = z;
    float pr = gauss ? (-L2PI_F - 0.5f * z * z + ld) : (-2.f * L2PI_F + ld);
    probout[r] = pr;
  }
}

// ---------------------------------------------------------------------------
// Weight prep: fp32 -> bf16, transposed to [N][K] with K/N padding.
// ---------------------------------------------------------------------------
__global__ void prep_net(const float* __restrict__ W1, const float* __restrict__ b1,
                         const float* __restrict__ W2, const float* __restrict__ b2,
                         const float* __restrict__ W3, const float* __restrict__ b3,
                         int din,
                         unsigned short* __restrict__ W1t,
                         unsigned short* __restrict__ W2t,
                         unsigned short* __restrict__ W3t,
                         float* __restrict__ b1o,
                         float* __restrict__ b2o,
                         float* __restrict__ b3o) {
  const int NW1 = 256 * 96, NW2 = 256 * 256, NW3 = 112 * 256;
  int i = blockIdx.x * blockDim.x + threadIdx.x;
  if (i < NW1) {
    int n = i / 96, k = i % 96;
    W1t[i] = f2bf(k < din ? W1[(size_t)k * 256 + n] : 0.f);
  } else if ((i -= NW1) < NW2) {
    int n = i / 256, k = i % 256;
    W2t[i] = f2bf(W2[(size_t)k * 256 + n]);
  } else if ((i -= NW2) < NW3) {
    int n = i / 256, k = i % 256;
    W3t[i] = f2bf(n < 97 ? W3[(size_t)k * 97 + n] : 0.f);
  } else if ((i -= NW3) < 256) {
    b1o[i] = b1[i];
  } else if ((i -= 256) < 256) {
    b2o[i] = b2[i];
  } else if ((i -= 256) < 112) {
    b3o[i] = (i < 97) ? b3[i] : 0.f;
  }
}

// ---------------------------------------------------------------------------
// Final reduce: out[b] = sum_p p1+p2+p3 + mask*pm
// ---------------------------------------------------------------------------
__global__ void reduce_kernel(const float* __restrict__ p1, const float* __restrict__ p2,
                              const float* __restrict__ p3, const float* __restrict__ pm,
                              const unsigned char* __restrict__ mask,
                              float* __restrict__ out, int Bn) {
  int b = blockIdx.x * blockDim.x + threadIdx.x;
  if (b >= Bn) return;
  float s = 0.f;
  for (int p = 0; p < 16; ++p) {
    int r = b * 16 + p;
    s += p1[r] + p2[r] + p3[r];
    if (mask[r]) s += pm[r];
  }
  out[b] = s;
}

// ---------------------------------------------------------------------------
// Host launcher
// ---------------------------------------------------------------------------
extern "C" void kernel_launch(void* const* d_in, const int* in_sizes, int n_in,
                              void* d_out, int out_size, void* d_ws, size_t ws_size,
                              hipStream_t stream) {
  (void)in_sizes; (void)n_in; (void)out_size; (void)ws_size;
  const int Bn = 8192, P = 16, R = Bn * P;

  const float* x = (const float*)d_in[0];
  const float* c = (const float*)d_in[1];
  const unsigned char* mask = (const unsigned char*)d_in[2];
  // params flattened in dict order: net_1, net_2, net_3, net_m; each W1,b1,W2,b2,W3,b3

  char* ws = (char*)d_ws;
  size_t off = 0;
  auto alloc = [&](size_t bytes) -> void* {
    void* p = ws + off;
    off = (off + bytes + 255) & ~(size_t)255;
    return p;
  };

  unsigned short *W1t[4], *W2t[4], *W3t[4];
  float *b1o[4], *b2o[4], *b3o[4];
  for (int n = 0; n < 4; ++n) {
    W1t[n] = (unsigned short*)alloc(256 * 96 * 2);
    W2t[n] = (unsigned short*)alloc(256 * 256 * 2);
    W3t[n] = (unsigned short*)alloc(112 * 256 * 2);
    b1o[n] = (float*)alloc(256 * 4);
    b2o[n] = (float*)alloc(256 * 4);
    b3o[n] = (float*)alloc(112 * 4);
  }
  float* z3 = (float*)alloc((size_t)R * 4);
  float* z2 = (float*)alloc((size_t)R * 4);
  float* pr[4];
  for (int n = 0; n < 4; ++n) pr[n] = (float*)alloc((size_t)R * 4);

  // Weight prep (net index n: 0=net_1, 1=net_2, 2=net_3, 3=net_m)
  const int prepWork = 256 * 96 + 256 * 256 + 112 * 256 + 256 + 256 + 112;
  const int prepGrid = (prepWork + 255) / 256;
  for (int n = 0; n < 4; ++n) {
    const float* W1 = (const float*)d_in[3 + 6 * n + 0];
    const float* b1 = (const float*)d_in[3 + 6 * n + 1];
    const float* W2 = (const float*)d_in[3 + 6 * n + 2];
    const float* b2 = (const float*)d_in[3 + 6 * n + 3];
    const float* W3 = (const float*)d_in[3 + 6 * n + 4];
    const float* b3 = (const float*)d_in[3 + 6 * n + 5];
    int din = (n == 3) ? 67 : 66;
    prep_net<<<prepGrid, 256, 0, stream>>>(W1, b1, W2, b2, W3, b3, din,
                                           W1t[n], W2t[n], W3t[n],
                                           b1o[n], b2o[n], b3o[n]);
  }

  const int G = R / MROWS;
  // net_3: in = [x0, x1, c], coord = x2, bound 1, periodic, z3 out
  mlp_rqs_kernel<<<G, 256, 0, stream>>>(c, x + 0, 4, x + 1, 4, nullptr, 0, 2,
                                        x + 2, 4,
                                        W1t[2], W2t[2], W3t[2], b1o[2], b2o[2], b3o[2],
                                        1.0f, 1, 0, z3, pr[2]);
  // net_m: in = [x0, x1, x2, c], coord = x3, bound 10
  mlp_rqs_kernel<<<G, 256, 0, stream>>>(c, x + 0, 4, x + 1, 4, x + 2, 4, 3,
                                        x + 3, 4,
                                        W1t[3], W2t[3], W3t[3], b1o[3], b2o[3], b3o[3],
                                        10.0f, 0, 1, nullptr, pr[3]);
  // net_2: in = [x0, z3, c], coord = x1, bound 1, z2 out
  mlp_rqs_kernel<<<G, 256, 0, stream>>>(c, x + 0, 4, z3, 1, nullptr, 0, 2,
                                        x + 1, 4,
                                        W1t[1], W2t[1], W3t[1], b1o[1], b2o[1], b3o[1],
                                        1.0f, 0, 0, z2, pr[1]);
  // net_1: in = [z2, z3, c], coord = x0, bound 10
  mlp_rqs_kernel<<<G, 256, 0, stream>>>(c, z2, 1, z3, 1, nullptr, 0, 2,
                                        x + 0, 4,
                                        W1t[0], W2t[0], W3t[0], b1o[0], b2o[0], b3o[0],
                                        10.0f, 0, 1, nullptr, pr[0]);

  reduce_kernel<<<(Bn + 255) / 256, 256, 0, stream>>>(pr[0], pr[1], pr[2], pr[3],
                                                      mask, (float*)d_out, Bn);
}
